// Net_1236950581989
// MI455X (gfx1250) — compile-verified, matrix operand order checked by reading, not run on hardware
//
#include <hip/hip_runtime.h>
#include <math.h>

// ---------------------------------------------------------------------------
// 2-layer GAT for MI455X (gfx1250, wave32).
// GEMMs via v_wmma_f32_16x16x32_f16 (fp32 accumulate, A converted f32->f16
// in-register). One wave owns a 16-row tile and sweeps ALL column tiles per
// K-step so the dominant A stream (x, 409.6 MB fp32) is read exactly once.
// Edge softmax/aggregation via global float atomics.
// ---------------------------------------------------------------------------

typedef __attribute__((ext_vector_type(16))) _Float16 v16h;
typedef __attribute__((ext_vector_type(8)))  _Float16 v8h;
typedef __attribute__((ext_vector_type(8)))  float    v8f;
typedef __attribute__((ext_vector_type(4)))  float    v4f;

#define GAT_N     100000
#define GAT_FIN   1024
#define GAT_H1    8
#define GAT_C1    8
#define GAT_D1    64      // H1*C1
#define GAT_COUT  40
#define GAT_COUTP 48      // padded to 3 x 16 WMMA column tiles
#define NEG_SLOPE 0.2f

// ---------------------------------------------------------------------------
// helpers
// ---------------------------------------------------------------------------
__device__ __forceinline__ float leaky(float x) {
    return x > 0.f ? x : NEG_SLOPE * x;
}

// atomic float max via int-max (v>=0) / uint-min (v<0); correct with -inf init
__device__ __forceinline__ void atomicMaxF(float* addr, float v) {
    if (v >= 0.f) atomicMax((int*)addr, __float_as_int(v));
    else          atomicMin((unsigned int*)addr, __float_as_uint(v));
}

__global__ void k_fill(float* __restrict__ p, long long cnt, float v) {
    long long i = (long long)blockIdx.x * blockDim.x + threadIdx.x;
    if (i < cnt) p[i] = v;
}

// ---------------------------------------------------------------------------
// weight transposes to f16 [col][K] so B fragments are contiguous per lane
// ---------------------------------------------------------------------------
__global__ void k_prep_w1(const float* __restrict__ W1, _Float16* __restrict__ W1t) {
    int i = blockIdx.x * blockDim.x + threadIdx.x;       // 64*1024
    if (i >= GAT_D1 * GAT_FIN) return;
    int c = i >> 10, k = i & (GAT_FIN - 1);
    W1t[(size_t)c * GAT_FIN + k] = (_Float16)W1[(size_t)k * GAT_D1 + c];
}

__global__ void k_prep_w2(const float* __restrict__ W2, _Float16* __restrict__ W2t) {
    int i = blockIdx.x * blockDim.x + threadIdx.x;       // 48*64
    if (i >= GAT_COUTP * GAT_D1) return;
    int c = i / GAT_D1, k = i % GAT_D1;
    float v = (c < GAT_COUT) ? W2[(size_t)k * GAT_COUT + c] : 0.f;
    W2t[(size_t)c * GAT_D1 + k] = (_Float16)v;
}

// ---------------------------------------------------------------------------
// WMMA GEMM: C[M x NT*16] = cvt_f16(A_f32[M x K]) * Bt^T
//   A  : row-major f32, lda == K
//   Bt : f16, [col][K] (pre-transposed weights)
//   C  : row-major f32, ldc == NT*16
// One wave per 16-row tile; the wave sweeps NT column tiles per K-step so the
// A fragment is fetched once and feeds NT back-to-back v_wmma issues.
// blockDim = (32, WPB); grid.x = Mtiles / WPB.
// ---------------------------------------------------------------------------
template <int NT>
__global__ void k_gemm_wmma(const float* __restrict__ A,
                            const _Float16* __restrict__ Bt,
                            float* __restrict__ C, int K, int ldc) {
    const int lane    = threadIdx.x;                    // 0..31, full wave
    const int tM      = blockIdx.x * blockDim.y + threadIdx.y;
    const int halfSel = lane >> 4;                      // 0: lanes 0-15
    const int l15     = lane & 15;
    const int koff    = halfSel * 8;

    const float*    aRow  = A  + (size_t)(tM * 16 + l15) * K;
    const _Float16* bBase = Bt + (size_t)l15 * K;       // + t*16*K per col tile

    v8f acc[NT];
#pragma unroll
    for (int t = 0; t < NT; ++t) acc[t] = (v8f){};

    for (int k0 = 0; k0 < K; k0 += 32) {
        const int ka = k0 + koff;
        __builtin_prefetch(aRow + ka + 64, 0, 1);       // global_prefetch_b8
        // A fragment: K = ka..ka+7 then ka+16..ka+23 (ISA 16-bit A layout)
        v4f a0 = *(const v4f*)(aRow + ka);
        v4f a1 = *(const v4f*)(aRow + ka + 4);
        v4f a2 = *(const v4f*)(aRow + ka + 16);
        v4f a3 = *(const v4f*)(aRow + ka + 20);
        v16h av;
#pragma unroll
        for (int i = 0; i < 4; ++i) {
            av[i]      = (_Float16)a0[i];
            av[4 + i]  = (_Float16)a1[i];
            av[8 + i]  = (_Float16)a2[i];
            av[12 + i] = (_Float16)a3[i];
        }
#pragma unroll
        for (int t = 0; t < NT; ++t) {
            const _Float16* bRow = bBase + (size_t)t * 16 * K;
            v8h b0 = *(const v8h*)(bRow + ka);
            v8h b1 = *(const v8h*)(bRow + ka + 16);
            v16h bv;
#pragma unroll
            for (int i = 0; i < 8; ++i) { bv[i] = b0[i]; bv[8 + i] = b1[i]; }
            acc[t] = __builtin_amdgcn_wmma_f32_16x16x32_f16(
                /*neg_a=*/false, av, /*neg_b=*/false, bv,
                /*c_mod=*/(short)0, acc[t], /*reuse_a=*/false, /*reuse_b=*/false);
        }
    }
    // C/D layout: VGPR v -> row (v + 8*halfSel), col = l15 (+16 per tile)
#pragma unroll
    for (int t = 0; t < NT; ++t)
#pragma unroll
        for (int v = 0; v < 8; ++v)
            C[(size_t)(tM * 16 + halfSel * 8 + v) * ldc + t * 16 + l15] = acc[t][v];
}

// ---------------------------------------------------------------------------
// per-node attention logits
// ---------------------------------------------------------------------------
__global__ void k_logits1(const float* __restrict__ xw,
                          const float* __restrict__ aS, const float* __restrict__ aD,
                          float* __restrict__ als, float* __restrict__ ald, int N) {
    int n = blockIdx.x * blockDim.x + threadIdx.x;
    if (n >= N) return;
    const float* r = xw + (size_t)n * GAT_D1;
#pragma unroll
    for (int h = 0; h < GAT_H1; ++h) {
        float s = 0.f, t = 0.f;
#pragma unroll
        for (int c = 0; c < GAT_C1; ++c) {
            float v = r[h * GAT_C1 + c];
            s += v * aS[h * GAT_C1 + c];
            t += v * aD[h * GAT_C1 + c];
        }
        als[(size_t)n * GAT_H1 + h] = s;
        ald[(size_t)n * GAT_H1 + h] = t;
    }
}

__global__ void k_logits2(const float* __restrict__ hw,
                          const float* __restrict__ aS, const float* __restrict__ aD,
                          float* __restrict__ als, float* __restrict__ ald, int N) {
    int n = blockIdx.x * blockDim.x + threadIdx.x;
    if (n >= N) return;
    const float* r = hw + (size_t)n * GAT_COUTP;
    float s = 0.f, t = 0.f;
#pragma unroll
    for (int c = 0; c < GAT_COUT; ++c) {
        float v = r[c];
        s += v * aS[c];
        t += v * aD[c];
    }
    als[n] = s;
    ald[n] = t;
}

// ---------------------------------------------------------------------------
// edge passes (E directed edges + N self loops)
// ---------------------------------------------------------------------------
__device__ __forceinline__ void edge_sd(const int* ei, int E, int e, int& s, int& d) {
    if (e < E) { s = ei[e]; d = ei[E + e]; }
    else       { s = d = e - E; }
}

template <int H>
__global__ void k_edge_max(const int* __restrict__ ei, int E, int N,
                           const float* __restrict__ als, const float* __restrict__ ald,
                           float* __restrict__ m) {
    int e = blockIdx.x * blockDim.x + threadIdx.x;
    if (e >= E + N) return;
    int s, d; edge_sd(ei, E, e, s, d);
#pragma unroll
    for (int h = 0; h < H; ++h) {
        float sc = leaky(als[(size_t)s * H + h] + ald[(size_t)d * H + h]);
        atomicMaxF(&m[(size_t)d * H + h], sc);
    }
}

template <int H>
__global__ void k_edge_sum(const int* __restrict__ ei, int E, int N,
                           const float* __restrict__ als, const float* __restrict__ ald,
                           const float* __restrict__ m, float* __restrict__ dn) {
    int e = blockIdx.x * blockDim.x + threadIdx.x;
    if (e >= E + N) return;
    int s, d; edge_sd(ei, E, e, s, d);
#pragma unroll
    for (int h = 0; h < H; ++h) {
        float sc = leaky(als[(size_t)s * H + h] + ald[(size_t)d * H + h]);
        atomicAdd(&dn[(size_t)d * H + h], expf(sc - m[(size_t)d * H + h]));
    }
}

template <int H, int C>
__global__ void k_edge_agg(const int* __restrict__ ei, int E, int N,
                           const float* __restrict__ als, const float* __restrict__ ald,
                           const float* __restrict__ m, const float* __restrict__ dn,
                           const float* __restrict__ feat, int ldF,
                           float* __restrict__ agg, int ldO) {
    int e = blockIdx.x * blockDim.x + threadIdx.x;
    if (e >= E + N) return;
    int s, d; edge_sd(ei, E, e, s, d);
    float alpha[H];
#pragma unroll
    for (int h = 0; h < H; ++h) {
        float sc = leaky(als[(size_t)s * H + h] + ald[(size_t)d * H + h]);
        float p  = expf(sc - m[(size_t)d * H + h]);
        alpha[h] = p / (dn[(size_t)d * H + h] + 1e-16f);
    }
    const float* fr = feat + (size_t)s * ldF;
    float* orow     = agg  + (size_t)d * ldO;
    __builtin_prefetch(fr, 0, 1);
#pragma unroll
    for (int h = 0; h < H; ++h)
#pragma unroll
        for (int c = 0; c < C; ++c)
            atomicAdd(&orow[h * C + c], fr[h * C + c] * alpha[h]);
}

// ---------------------------------------------------------------------------
// epilogues
// ---------------------------------------------------------------------------
__global__ void k_elu_bias(float* __restrict__ h, const float* __restrict__ b,
                           long long total) {
    long long i = (long long)blockIdx.x * blockDim.x + threadIdx.x;
    if (i >= total) return;
    float v = h[i] + b[i & (GAT_D1 - 1)];
    h[i] = v > 0.f ? v : (expf(v) - 1.f);
}

__global__ void k_logsoftmax(const float* __restrict__ agg, const float* __restrict__ b,
                             float* __restrict__ out, int N) {
    int n = blockIdx.x * blockDim.x + threadIdx.x;
    if (n >= N) return;
    float v[GAT_COUT];
    float mx = -INFINITY;
#pragma unroll
    for (int c = 0; c < GAT_COUT; ++c) {
        v[c] = agg[(size_t)n * GAT_COUT + c] + b[c];
        mx = fmaxf(mx, v[c]);
    }
    float sum = 0.f;
#pragma unroll
    for (int c = 0; c < GAT_COUT; ++c) sum += expf(v[c] - mx);
    float lse = logf(sum) + mx;
#pragma unroll
    for (int c = 0; c < GAT_COUT; ++c)
        out[(size_t)n * GAT_COUT + c] = v[c] - lse;
}

// ---------------------------------------------------------------------------
// host launcher
// ---------------------------------------------------------------------------
extern "C" void kernel_launch(void* const* d_in, const int* in_sizes, int n_in,
                              void* d_out, int out_size, void* d_ws, size_t ws_size,
                              hipStream_t stream) {
    const float* x   = (const float*)d_in[0];
    const int*   ei  = (const int*)  d_in[1];
    const float* W1  = (const float*)d_in[2];
    const float* as1 = (const float*)d_in[3];
    const float* ad1 = (const float*)d_in[4];
    const float* b1  = (const float*)d_in[5];
    const float* W2  = (const float*)d_in[6];
    const float* as2 = (const float*)d_in[7];
    const float* ad2 = (const float*)d_in[8];
    const float* b2  = (const float*)d_in[9];

    const int N  = GAT_N;
    const int E  = in_sizes[1] / 2;
    const int ET = E + N;

    // workspace layout (floats; all regions 16B aligned)
    float* p     = (float*)d_ws;
    float* xw    = p;                 p += (size_t)N * GAT_D1;     // [N][64]
    float* agg1  = p;                 p += (size_t)N * GAT_D1;     // [N][64] -> h after ELU
    float* alS1  = p;                 p += (size_t)N * GAT_H1;
    float* alD1  = p;                 p += (size_t)N * GAT_H1;
    float* m1    = p;                 p += (size_t)N * GAT_H1;
    float* dn1   = p;                 p += (size_t)N * GAT_H1;
    float* hw    = p;                 p += (size_t)N * GAT_COUTP;  // [N][48]
    float* alS2  = p;                 p += (size_t)N;
    float* alD2  = p;                 p += (size_t)N;
    float* m2    = p;                 p += (size_t)N;
    float* dn2   = p;                 p += (size_t)N;
    float* agg2  = p;                 p += (size_t)N * GAT_COUT;   // [N][40]
    _Float16* W1t = (_Float16*)p;                                  // [64][1024]
    _Float16* W2t = W1t + (size_t)GAT_D1 * GAT_FIN;                // [48][64]

    const int TB = 256;
    auto nb = [](long long c, int b) { return (unsigned)((c + b - 1) / b); };

    // weight prep
    k_prep_w1<<<nb(GAT_D1 * GAT_FIN, TB), TB, 0, stream>>>(W1, W1t);
    k_prep_w2<<<nb(GAT_COUTP * GAT_D1, TB), TB, 0, stream>>>(W2, W2t);

    // re-init accumulators every call (harness does not re-poison)
    k_fill<<<nb((long long)N * GAT_H1, TB), TB, 0, stream>>>(m1, (long long)N * GAT_H1, -INFINITY);
    k_fill<<<nb((long long)N * GAT_H1, TB), TB, 0, stream>>>(dn1, (long long)N * GAT_H1, 0.f);
    k_fill<<<nb((long long)N * GAT_D1, TB), TB, 0, stream>>>(agg1, (long long)N * GAT_D1, 0.f);
    k_fill<<<nb(N, TB), TB, 0, stream>>>(m2, N, -INFINITY);
    // dn2 and agg2 are contiguous: zero in one shot
    k_fill<<<nb((long long)N * (1 + GAT_COUT), TB), TB, 0, stream>>>(dn2, (long long)N * (1 + GAT_COUT), 0.f);

    // ---- layer 1 ----  (Mtiles = 6250 = 3125 blocks x 2 waves)
    k_gemm_wmma<GAT_D1 / 16><<<dim3(N / 32), dim3(32, 2), 0, stream>>>(x, W1t, xw, GAT_FIN, GAT_D1);
    k_logits1<<<nb(N, TB), TB, 0, stream>>>(xw, as1, ad1, alS1, alD1, N);
    k_edge_max<GAT_H1><<<nb(ET, TB), TB, 0, stream>>>(ei, E, N, alS1, alD1, m1);
    k_edge_sum<GAT_H1><<<nb(ET, TB), TB, 0, stream>>>(ei, E, N, alS1, alD1, m1, dn1);
    k_edge_agg<GAT_H1, GAT_C1><<<nb(ET, TB), TB, 0, stream>>>(ei, E, N, alS1, alD1, m1, dn1,
                                                              xw, GAT_D1, agg1, GAT_D1);
    k_elu_bias<<<nb((long long)N * GAT_D1, TB), TB, 0, stream>>>(agg1, b1, (long long)N * GAT_D1);

    // ---- layer 2 ----
    k_gemm_wmma<GAT_COUTP / 16><<<dim3(N / 32), dim3(32, 2), 0, stream>>>(agg1, W2t, hw, GAT_D1, GAT_COUTP);
    k_logits2<<<nb(N, TB), TB, 0, stream>>>(hw, as2, ad2, alS2, alD2, N);
    k_edge_max<1><<<nb(ET, TB), TB, 0, stream>>>(ei, E, N, alS2, alD2, m2);
    k_edge_sum<1><<<nb(ET, TB), TB, 0, stream>>>(ei, E, N, alS2, alD2, m2, dn2);
    k_edge_agg<1, GAT_COUT><<<nb(ET, TB), TB, 0, stream>>>(ei, E, N, alS2, alD2, m2, dn2,
                                                           hw, GAT_COUTP, agg2, GAT_COUT);

    // ---- log_softmax ----
    k_logsoftmax<<<nb(N, TB), TB, 0, stream>>>(agg2, b2, (float*)d_out, N);
}
